// AdvancedTransformer_44942537785737
// MI455X (gfx1250) — matrix operand order repaired and checked by reading, not compiled, not run on hardware
//
#include <hip/hip_runtime.h>
#include <hip/hip_bf16.h>
#include <math.h>

typedef __attribute__((ext_vector_type(16))) _Float16 v16h;
typedef __attribute__((ext_vector_type(8)))  float    v8f;
typedef int v4i __attribute__((vector_size(16)));

#define DIM    1024
#define HEADS  16
#define HDK    64
#define LAYERS 6
#define FFDIM  4096
#define BATCH  8
#define SEQ    1024
#define ROWS   (BATCH*SEQ)   // 8192

// ---------------------------------------------------------------------------
// Async global->LDS staging (CDNA5 GLOBAL_LOAD_ASYNC_TO_LDS_B128, ASYNCcnt).
// Builtin signature: (v4i AS(1)*, v4i AS(3)*, imm offset, imm cpol).
// Guarded: falls back to a plain LDS copy if the builtin is absent.
// LDS flat address low 32 bits == LDS offset (aperture rules, ISA 10.2).
// ---------------------------------------------------------------------------
#if __has_builtin(__builtin_amdgcn_global_load_async_to_lds_b128)
#define HAVE_ASYNC_LDS 1
#else
#define HAVE_ASYNC_LDS 0
#endif

__device__ __forceinline__ void async_copy_b128(void* lds_dst, const void* gsrc) {
#if HAVE_ASYNC_LDS
  __attribute__((address_space(1))) v4i* g =
      (__attribute__((address_space(1))) v4i*)(uintptr_t)gsrc;
  __attribute__((address_space(3))) v4i* l =
      (__attribute__((address_space(3))) v4i*)(unsigned)(uintptr_t)lds_dst;
  __builtin_amdgcn_global_load_async_to_lds_b128(g, l, 0, 0);
#else
  *(uint4*)lds_dst = *(const uint4*)gsrc;
#endif
}

__device__ __forceinline__ void async_wait() {
#if HAVE_ASYNC_LDS
  asm volatile("s_wait_asynccnt 0" ::: "memory");
#endif
}

// ---------------------------------------------------------------------------
// WMMA fragment loaders, per CDNA5 ISA 7.12.2 VGPR layouts (wave32).
// A (16x32 f16): lane m=l&15, half=l>>4; element e -> k = (e<8 ? e : e+8) + 8*half
//   => two contiguous 8-half (16B) chunks at k = 8*half and k = 8*half+16.
// B (32x16 f16): lane n=l&15, half=l>>4; element e -> k = (e<8 ? e : e+8) + 8*half
//   => two contiguous 8-half (16B) chunks at k = 8*half and k = 8*half+16.
// C/D (16x16 f32, 8 VGPRs): VGPR r holds row M = r + 8*half, col N = l&15.
// ---------------------------------------------------------------------------
union Frag16 { v16h v; uint4 q[2]; };

__device__ __forceinline__ v16h ld_frag_a(const _Float16* base, int lane, int ld) {
  Frag16 f;
  int m = lane & 15, hl = lane >> 4;
  const _Float16* p = base + m * ld + hl * 8;
  f.q[0] = *(const uint4*)(p);
  f.q[1] = *(const uint4*)(p + 16);
  return f.v;
}

__device__ __forceinline__ v16h ld_frag_b(const _Float16* base, int lane, int ld) {
  // base points at n-major storage: base[n*ld + k]
  Frag16 f;
  int n = lane & 15, hl = lane >> 4;
  const _Float16* p = base + n * ld + hl * 8;
  f.q[0] = *(const uint4*)(p);
  f.q[1] = *(const uint4*)(p + 16);
  return f.v;
}

// ---------------------------------------------------------------------------
// GEMM: C[M,N] = act(A16[M,K] @ Bw32[K,N] + bias)
//  A: f16 row-major, staged by async global->LDS b128 copies.
//  B: fp32 weights, converted to f16 while staging (n-major in LDS).
//  Block tile 128x128, BK=32. 8 waves; each wave owns 32x64 (2x4 WMMA tiles).
// ---------------------------------------------------------------------------
template<bool BIAS, bool RELU, bool OUT16>
__global__ __launch_bounds__(256)
void gemm_f16w(const _Float16* __restrict__ A, const float* __restrict__ B,
               const float* __restrict__ bias,
               float* __restrict__ C32, _Float16* __restrict__ C16,
               int M, int N, int K)
{
  (void)M;
  constexpr int BM = 128, BN = 128, BK = 32, LDT = BK + 8; // 40 halves = 80B stride
  __shared__ __align__(16) _Float16 As[BM * LDT];
  __shared__ __align__(16) _Float16 Bs[BN * LDT];          // transposed: [n][k]

  const int tid  = threadIdx.x;
  const int lane = tid & 31, wave = tid >> 5;
  const int wm = wave >> 1, wn = wave & 1;                 // 4x2 wave grid
  const int m0 = blockIdx.y * BM, n0 = blockIdx.x * BN;

  v8f acc[2][4] = {};

  for (int kk = 0; kk < K; kk += BK) {
    __syncthreads();
    // A tile: 128 rows x 32 halves = 512 chunks of 8 halves (16B), async path
    #pragma unroll
    for (int i = 0; i < 2; ++i) {
      int c   = tid + i * 256;
      int row = c >> 2, kc = (c & 3) * 8;
      async_copy_b128(As + row * LDT + kc,
                      A + (size_t)(m0 + row) * K + kk + kc);
    }
    // B tile: 32 k-rows x 128 n-cols fp32 -> f16, transposed to [n][k]
    #pragma unroll
    for (int i = 0; i < 4; ++i) {
      int f   = tid + i * 256;               // 1024 float4 chunks
      int row = f >> 5, c4 = (f & 31) * 4;
      float4 d = *(const float4*)(B + (size_t)(kk + row) * N + n0 + c4);
      Bs[(c4 + 0) * LDT + row] = (_Float16)d.x;
      Bs[(c4 + 1) * LDT + row] = (_Float16)d.y;
      Bs[(c4 + 2) * LDT + row] = (_Float16)d.z;
      Bs[(c4 + 3) * LDT + row] = (_Float16)d.w;
    }
    if (kk + BK < K)  // pull next weight tile toward L2/L0
      __builtin_prefetch(B + (size_t)(kk + BK) * N + n0 + (tid & 31) * 4, 0, 1);
    async_wait();
    __syncthreads();

    v16h a[2], b[4];
    #pragma unroll
    for (int i = 0; i < 2; ++i) a[i] = ld_frag_a(As + (wm * 32 + i * 16) * LDT, lane, LDT);
    #pragma unroll
    for (int j = 0; j < 4; ++j) b[j] = ld_frag_b(Bs + (wn * 64 + j * 16) * LDT, lane, LDT);
    #pragma unroll
    for (int i = 0; i < 2; ++i)
      #pragma unroll
      for (int j = 0; j < 4; ++j)
        acc[i][j] = __builtin_amdgcn_wmma_f32_16x16x32_f16(
            false, a[i], false, b[j], (short)0, acc[i][j], false, false);
  }

  const int hl = lane >> 4, nl = lane & 15;
  #pragma unroll
  for (int i = 0; i < 2; ++i)
    #pragma unroll
    for (int j = 0; j < 4; ++j) {
      int col = n0 + wn * 64 + j * 16 + nl;
      float bv = BIAS ? bias[col] : 0.f;
      #pragma unroll
      for (int r = 0; r < 8; ++r) {
        int row = m0 + wm * 32 + i * 16 + r + 8 * hl;
        float v = acc[i][j][r] + bv;
        if (RELU) v = v > 0.f ? v : 0.f;
        if (OUT16) C16[(size_t)row * N + col] = (_Float16)v;
        else       C32[(size_t)row * N + col] = v;
      }
    }
}

// ---------------------------------------------------------------------------
// Flash attention: block = one (b,h) and 64 q rows (4 waves x 16 rows).
// Loops over 32-key tiles; online softmax done in WMMA C-layout.
// K tile staged via async global->LDS; V staged manually (needs transpose).
// ---------------------------------------------------------------------------
__global__ __launch_bounds__(128)
void attn_kernel(const _Float16* __restrict__ Q, const _Float16* __restrict__ Kg,
                 const _Float16* __restrict__ Vg, const int* __restrict__ tok,
                 _Float16* __restrict__ O)
{
  constexpr int LDK = 72;   // Ks row stride (halves): 144B, 16B multiple
  constexpr int LDV = 40;   // VsT row stride: 80B
  __shared__ __align__(16) _Float16 Ks [32 * LDK];   // [key][d]
  __shared__ __align__(16) _Float16 VsT[64 * LDV];   // [d][key]
  __shared__ __align__(16) _Float16 Ps [4 * 16 * 32];
  __shared__ float msk[32];

  const int tid = threadIdx.x, lane = tid & 31, wave = tid >> 5;
  const int qt = blockIdx.x, h = blockIdx.y, b = blockIdx.z;
  const int hOff = h * HDK;
  const size_t rowBase = (size_t)b * SEQ;
  const int qRow0 = qt * 64 + wave * 16;
  const int nl = lane & 15, hl = lane >> 4;
  const float scale = 0.125f;  // 1/sqrt(64)

  // Q fragments (A, 16x32, K-dim = head dim split 0..31 / 32..63)
  v16h qa[2];
  {
    const _Float16* qp = Q + (rowBase + qRow0) * DIM + hOff;
    qa[0] = ld_frag_a(qp,      lane, DIM);
    qa[1] = ld_frag_a(qp + 32, lane, DIM);
  }

  float mPrev[8], lsum[8];
  #pragma unroll
  for (int r = 0; r < 8; ++r) { mPrev[r] = -3.0e38f; lsum[r] = 0.f; }
  v8f acc[4] = {};

  for (int kk = 0; kk < SEQ; kk += 32) {
    __syncthreads();
    #pragma unroll
    for (int i = 0; i < 2; ++i) {
      int c = tid + i * 128;                 // 256 chunks of 8 halves
      int key = c >> 3, d0 = (c & 7) * 8;
      async_copy_b128(Ks + key * LDK + d0,
                      Kg + (rowBase + kk + key) * DIM + hOff + d0);
      uint4 dv = *(const uint4*)(Vg + (rowBase + kk + key) * DIM + hOff + d0);
      const _Float16* hv = (const _Float16*)&dv;
      #pragma unroll
      for (int j = 0; j < 8; ++j) VsT[(d0 + j) * LDV + key] = hv[j];
    }
    if (tid < 32) msk[tid] = (tok[rowBase + kk + tid] != 0) ? 0.f : -1e9f;
    async_wait();
    __syncthreads();

    // scores: two 16-key subtiles, K-dim 64 (2 WMMAs each)
    v8f sc[2] = {};
    #pragma unroll
    for (int j = 0; j < 2; ++j) {
      sc[j] = __builtin_amdgcn_wmma_f32_16x16x32_f16(false, qa[0], false,
                  ld_frag_b(Ks + (j * 16) * LDK, lane, LDK), (short)0, sc[j], false, false);
      sc[j] = __builtin_amdgcn_wmma_f32_16x16x32_f16(false, qa[1], false,
                  ld_frag_b(Ks + (j * 16) * LDK + 32, lane, LDK), (short)0, sc[j], false, false);
    }

    const float mv0 = msk[nl], mv1 = msk[16 + nl];
    _Float16* pw = Ps + wave * (16 * 32);
    #pragma unroll
    for (int r = 0; r < 8; ++r) {
      float s0 = sc[0][r] * scale + mv0;
      float s1 = sc[1][r] * scale + mv1;
      float mx = fmaxf(s0, s1);
      #pragma unroll
      for (int d = 1; d < 16; d <<= 1) mx = fmaxf(mx, __shfl_xor(mx, d, 32));
      float mN = fmaxf(mPrev[r], mx);
      float ef = __expf(mPrev[r] - mN);
      float p0 = __expf(s0 - mN), p1 = __expf(s1 - mN);
      float rs = p0 + p1;
      #pragma unroll
      for (int d = 1; d < 16; d <<= 1) rs += __shfl_xor(rs, d, 32);
      lsum[r] = lsum[r] * ef + rs;
      mPrev[r] = mN;
      #pragma unroll
      for (int dn = 0; dn < 4; ++dn) acc[dn][r] *= ef;
      int prow = r + 8 * hl;                 // C-layout row for this lane
      pw[prow * 32 + nl]      = (_Float16)p0;
      pw[prow * 32 + 16 + nl] = (_Float16)p1;
    }
    __syncthreads();

    v16h pa = ld_frag_a(pw, lane, 32);       // P as A fragment (16x32)
    #pragma unroll
    for (int dn = 0; dn < 4; ++dn)
      acc[dn] = __builtin_amdgcn_wmma_f32_16x16x32_f16(false, pa, false,
                    ld_frag_b(VsT + (dn * 16) * LDV, lane, LDV), (short)0, acc[dn], false, false);
  }

  #pragma unroll
  for (int dn = 0; dn < 4; ++dn)
    #pragma unroll
    for (int r = 0; r < 8; ++r) {
      int row = qRow0 + r + 8 * hl;
      float o = acc[dn][r] / lsum[r];
      O[(rowBase + row) * DIM + hOff + dn * 16 + nl] = (_Float16)o;
    }
}

// ---------------------------------------------------------------------------
// Embedding gather + sqrt(D) scale + sinusoidal positional encoding.
// ---------------------------------------------------------------------------
__global__ __launch_bounds__(256)
void embed_kernel(const int* __restrict__ tok, const float* __restrict__ emb,
                  float* __restrict__ x, _Float16* __restrict__ xh)
{
  int row = blockIdx.x;
  int s = row & (SEQ - 1);
  int t = tok[row];
  const float* e = emb + (size_t)t * DIM;
  const float nlog = -logf(10000.f) / (float)DIM;
  for (int d = threadIdx.x; d < DIM; d += 256) {
    float div = __expf(nlog * (float)(d & ~1));
    float ang = (float)s * div;
    float pe = (d & 1) ? __cosf(ang) : __sinf(ang);
    float v = e[d] * 32.0f + pe;
    x [(size_t)row * DIM + d] = v;
    xh[(size_t)row * DIM + d] = (_Float16)v;
  }
}

// ---------------------------------------------------------------------------
// Residual add + LayerNorm. FINAL: no residual, write fp32 out only.
// ---------------------------------------------------------------------------
template<bool FINAL>
__global__ __launch_bounds__(256)
void addln_kernel(float* __restrict__ x, const float* __restrict__ z,
                  const float* __restrict__ w, const float* __restrict__ bb,
                  _Float16* __restrict__ xh, float* __restrict__ out, float eps)
{
  __shared__ float red[256];
  const int row = blockIdx.x, tid = threadIdx.x;
  float vals[4];
  float s = 0.f;
  #pragma unroll
  for (int i = 0; i < 4; ++i) {
    int d = tid + i * 256;
    float v = x[(size_t)row * DIM + d];
    if (!FINAL) v += z[(size_t)row * DIM + d];
    vals[i] = v; s += v;
  }
  red[tid] = s; __syncthreads();
  for (int st = 128; st > 0; st >>= 1) { if (tid < st) red[tid] += red[tid + st]; __syncthreads(); }
  float mean = red[0] * (1.0f / DIM); __syncthreads();
  float vs = 0.f;
  #pragma unroll
  for (int i = 0; i < 4; ++i) { float d0 = vals[i] - mean; vs += d0 * d0; }
  red[tid] = vs; __syncthreads();
  for (int st = 128; st > 0; st >>= 1) { if (tid < st) red[tid] += red[tid + st]; __syncthreads(); }
  float inv = rsqrtf(red[0] * (1.0f / DIM) + eps);
  #pragma unroll
  for (int i = 0; i < 4; ++i) {
    int d = tid + i * 256;
    float y = (vals[i] - mean) * inv * w[d] + bb[d];
    if (FINAL) out[(size_t)row * DIM + d] = y;
    else { x[(size_t)row * DIM + d] = y; xh[(size_t)row * DIM + d] = (_Float16)y; }
  }
}

// ---------------------------------------------------------------------------
extern "C" void kernel_launch(void* const* d_in, const int* in_sizes, int n_in,
                              void* d_out, int out_size, void* d_ws, size_t ws_size,
                              hipStream_t stream)
{
  (void)in_sizes; (void)n_in; (void)out_size; (void)ws_size;
  const int*   tok  = (const int*)  d_in[0];
  const float* emb  = (const float*)d_in[1];
  const float* Wq   = (const float*)d_in[2];
  const float* Wk   = (const float*)d_in[3];
  const float* Wv   = (const float*)d_in[4];
  const float* Wo   = (const float*)d_in[5];
  const float* W1   = (const float*)d_in[6];
  const float* b1   = (const float*)d_in[7];
  const float* W2   = (const float*)d_in[8];
  const float* b2   = (const float*)d_in[9];
  const float* ln1w = (const float*)d_in[10];
  const float* ln1b = (const float*)d_in[11];
  const float* ln2w = (const float*)d_in[12];
  const float* ln2b = (const float*)d_in[13];
  const float* fnw  = (const float*)d_in[14];
  const float* fnb  = (const float*)d_in[15];
  float* out = (float*)d_out;

  char* ws = (char*)d_ws;
  size_t o = 0;
  float*    x     = (float*)   (ws + o); o += (size_t)ROWS * DIM * 4;
  float*    tmp   = (float*)   (ws + o); o += (size_t)ROWS * DIM * 4;
  _Float16* xh    = (_Float16*)(ws + o); o += (size_t)ROWS * DIM * 2;
  _Float16* q16   = (_Float16*)(ws + o); o += (size_t)ROWS * DIM * 2;
  _Float16* k16   = (_Float16*)(ws + o); o += (size_t)ROWS * DIM * 2;
  _Float16* v16   = (_Float16*)(ws + o); o += (size_t)ROWS * DIM * 2;
  _Float16* ctx16 = (_Float16*)(ws + o); o += (size_t)ROWS * DIM * 2;
  _Float16* ff16  = (_Float16*)(ws + o); o += (size_t)ROWS * FFDIM * 2;

  embed_kernel<<<ROWS, 256, 0, stream>>>(tok, emb, x, xh);

  const dim3 blk(256);
  const dim3 gProj(DIM / 128, ROWS / 128);     // (8, 64)
  const dim3 gFF1 (FFDIM / 128, ROWS / 128);   // (32, 64)
  const dim3 gAttn(SEQ / 64, HEADS, BATCH);

  for (int l = 0; l < LAYERS; ++l) {
    const float* wq = Wq + (size_t)l * DIM * DIM;
    const float* wk = Wk + (size_t)l * DIM * DIM;
    const float* wv = Wv + (size_t)l * DIM * DIM;
    const float* wo = Wo + (size_t)l * DIM * DIM;
    const float* w1 = W1 + (size_t)l * DIM * FFDIM;
    const float* w2 = W2 + (size_t)l * FFDIM * DIM;

    gemm_f16w<false,false,true ><<<gProj, blk, 0, stream>>>(xh, wq, nullptr, nullptr, q16, ROWS, DIM, DIM);
    gemm_f16w<false,false,true ><<<gProj, blk, 0, stream>>>(xh, wk, nullptr, nullptr, k16, ROWS, DIM, DIM);
    gemm_f16w<false,false,true ><<<gProj, blk, 0, stream>>>(xh, wv, nullptr, nullptr, v16, ROWS, DIM, DIM);

    attn_kernel<<<gAttn, 128, 0, stream>>>(q16, k16, v16, tok, ctx16);

    gemm_f16w<false,false,false><<<gProj, blk, 0, stream>>>(ctx16, wo, nullptr, tmp, nullptr, ROWS, DIM, DIM);
    addln_kernel<false><<<ROWS, 256, 0, stream>>>(x, tmp, ln1w + (size_t)l * DIM, ln1b + (size_t)l * DIM, xh, nullptr, 1e-6f);

    gemm_f16w<true ,true ,true ><<<gFF1 , blk, 0, stream>>>(xh, w1, b1 + (size_t)l * FFDIM, nullptr, ff16, ROWS, FFDIM, DIM);
    gemm_f16w<true ,false,false><<<gProj, blk, 0, stream>>>(ff16, w2, b2 + (size_t)l * DIM, tmp, nullptr, ROWS, DIM, FFDIM);
    addln_kernel<false><<<ROWS, 256, 0, stream>>>(x, tmp, ln2w + (size_t)l * DIM, ln2b + (size_t)l * DIM, xh, nullptr, 1e-6f);
  }

  addln_kernel<true><<<ROWS, 256, 0, stream>>>(x, nullptr, fnw, fnb, nullptr, out, 1e-5f);
}
